// SplineCNN_18829136626439
// MI455X (gfx1250) — matrix operand order repaired and checked by reading, not compiled, not run on hardware
//
#include <hip/hip_runtime.h>
#include <hip/hip_bf16.h>

typedef __bf16 bf16;
typedef __attribute__((ext_vector_type(16))) __bf16 v16bf;
typedef __attribute__((ext_vector_type(8)))  float  v8f;
typedef __attribute__((ext_vector_type(4)))  unsigned int u32x4;
typedef __attribute__((ext_vector_type(8)))  int          i32x8;

#define N_NODES 50000
#define N_EDGES 1600000
#define C 128
#define KS 5
#define K 25          // KS*KS spline slots
#define BN_EPS 1e-5f
#define NTILES (N_NODES / 16)   // 3125, N divisible by 16

// ---------------------------------------------------------------------------
// TDM helpers: DMA one 16 KB contiguous tile (4096 dwords) global -> LDS.
// D# per CDNA5 ISA ch.8: group0 {count=1, lds_addr, global_addr, type=2},
// group1 {data_size=4B, tensor_dim0=tile_dim0=4096, dim1=1, stride=4096}.
// Inline asm (portable across ROCm7.2 / amdgpu-toolchain builtin arity skew).
// ---------------------------------------------------------------------------
__device__ __forceinline__ void tdm_load_16kb(const bf16* gsrc, unsigned lds_byte_addr) {
    unsigned long long ga = (unsigned long long)(uintptr_t)gsrc;
    u32x4 g0;
    g0[0] = 1u;                                   // count=1, no gather/iterate
    g0[1] = lds_byte_addr;                        // lds_addr (bytes)
    g0[2] = (unsigned)ga;                         // global_addr[31:0]
    g0[3] = ((unsigned)(ga >> 32) & 0x01FFFFFFu)  // global_addr[56:32]
            | 0x80000000u;                        // type=2 ("image")
    i32x8 g1;
    g1[0] = 2 << 16;                              // data_size=2 -> 4-byte elems
    g1[1] = 4096 << 16;                           // tensor_dim0[15:0]=4096
    g1[2] = 1 << 16;                              // tensor_dim0 hi=0 | tensor_dim1=1
    g1[3] = 4096 << 16;                           // tile_dim0=4096
    g1[4] = 1;                                    // tile_dim1=1, tile_dim2=0
    g1[5] = 4096;                                 // tensor_dim0_stride=4096
    g1[6] = 0;
    g1[7] = 0;
    asm volatile("tensor_load_to_lds %0, %1" :: "s"(g0), "s"(g1) : "memory");
}

__device__ __forceinline__ void wait_tensorcnt0() {
#if __has_builtin(__builtin_amdgcn_s_wait_tensorcnt)
    __builtin_amdgcn_s_wait_tensorcnt(0);
#else
    asm volatile("s_wait_tensorcnt 0x0" ::: "memory");
#endif
}

// ---------------------------------------------------------------------------
// Embedding lookup -> bf16 feature matrix h[N,128]
// ---------------------------------------------------------------------------
__global__ void embed_kernel(const int* __restrict__ x,
                             const float* __restrict__ emb,
                             bf16* __restrict__ hbf) {
    int i = blockIdx.x * 256 + threadIdx.x;
    if (i >= N_NODES * C) return;
    int n = i >> 7, c = i & 127;
    hbf[i] = (bf16)emb[x[n] * C + c];
}

// ---------------------------------------------------------------------------
// Pack 128x128 f32 matrices into the CDNA5 WMMA B-matrix (32x16, K-major per
// half-wave) bf16 register layout: per matrix flat index
//   ((ctile*4 + q)*32 + lane)*16 + e   with
//   row = q*32 + (lane/16)*8 + (e%8) + (e/8)*16 ; col = ctile*16 + lane%16
// ---------------------------------------------------------------------------
__global__ void pack_b_kernel(const float* __restrict__ W,
                              bf16* __restrict__ Wp,
                              int nmat, int transpose) {
    int i = blockIdx.x * 256 + threadIdx.x;
    if (i >= nmat * 16384) return;
    int m     = i / 16384;
    int r     = i & 16383;
    int e     = r & 15;
    int lane  = (r >> 4) & 31;
    int q     = (r >> 9) & 3;
    int ctile = r >> 11;
    int row = q * 32 + (lane >> 4) * 8 + (e & 7) + (e >> 3) * 16;
    int col = ctile * 16 + (lane & 15);
    float v = transpose ? W[(size_t)m * 16384 + col * 128 + row]
                        : W[(size_t)m * 16384 + row * 128 + col];
    Wp[i] = (bf16)v;
}

// ---------------------------------------------------------------------------
// A fragment (16x32 bf16, ISA layout) from row-major h[N,128]:
// lanes 0-15 hold K {0-7,16-23}, lanes 16-31 hold K {8-15,24-31}.
// ---------------------------------------------------------------------------
__device__ __forceinline__ v16bf load_a_frag(const bf16* __restrict__ hbf,
                                             int rowA, int q, int lane) {
    const bf16* base = hbf + (size_t)rowA * C + q * 32 + ((lane >> 4) * 8);
    v16bf a;
#pragma unroll
    for (int j = 0; j < 8; ++j) { a[j] = base[j]; a[j + 8] = base[16 + j]; }
    return a;
}

// ---------------------------------------------------------------------------
// Transform GEMM: T[n,k,c] = sum_i h[n,i] * W[k,i,c]   (25 GEMMs fused)
// Block = 256 thr = 8 waves; blockIdx = 16-row node tile; wave = c-tile.
// Per-k weight tile (16 KB) double-buffered in LDS via TDM: DMA of tile k+1
// overlaps the WMMAs of tile k. All 8 ds_load_b128 for an iteration are
// issued before the 4-WMMA accumulation chain so the DS latency is hidden
// behind the first WMMA instead of being waited on 4x per iteration.
// ---------------------------------------------------------------------------
__global__ void __launch_bounds__(256)
gemm_transform_kernel(const bf16* __restrict__ hbf,
                      const bf16* __restrict__ Wp,
                      float* __restrict__ T) {
    __shared__ __align__(32) bf16 lds_w[2][8192];   // 2 x 16 KB

    int ntile = blockIdx.x;
    int ctile = threadIdx.x >> 5;
    int lane  = threadIdx.x & 31;
    int rowA  = ntile * 16 + (lane & 15);

    if (threadIdx.x == 0)
        tdm_load_16kb(Wp, (unsigned)(uintptr_t)&lds_w[0][0]);

    v16bf a[4];
#pragma unroll
    for (int q = 0; q < 4; ++q) a[q] = load_a_frag(hbf, rowA, q, lane);

    int colD  = ctile * 16 + (lane & 15);
    int rbase = ntile * 16 + (lane >> 4) * 8;

    for (int k = 0; k < K; ++k) {
        if (threadIdx.x < 32) wait_tensorcnt0();   // wave 0 owns the TDM ops
        __syncthreads();                           // buf[k&1] ready for all
        if (threadIdx.x == 0 && k + 1 < K)
            tdm_load_16kb(Wp + (size_t)(k + 1) * 16384,
                          (unsigned)(uintptr_t)&lds_w[(k + 1) & 1][0]);

        const bf16* wbuf = &lds_w[k & 1][0];
        v16bf b[4];
#pragma unroll
        for (int q = 0; q < 4; ++q)
            b[q] = *(const v16bf*)(wbuf + ((ctile * 4 + q) * 512 + lane * 16));

        v8f acc = {};
#pragma unroll
        for (int q = 0; q < 4; ++q)
            acc = __builtin_amdgcn_wmma_f32_16x16x32_bf16(
                      false, a[q], false, b[q], (short)0, acc, false, false);

#pragma unroll
        for (int j = 0; j < 8; ++j)
            T[(size_t)(rbase + j) * (K * C) + k * C + colD] = acc[j];
    }
}

// ---------------------------------------------------------------------------
// Row GEMM: out[n,c] = Cinit(n,c) + sum_i h[n,i]*B[i,c]
//   Cinit = bias[c] (+ agg[n,c]/max(deg[n],1) when agg != null)
// Fuses SplineConv mean-agg + root transform + bias; also the final linear.
// ---------------------------------------------------------------------------
__global__ void __launch_bounds__(256)
gemm_rowlin_kernel(const bf16* __restrict__ hbf,
                   const bf16* __restrict__ Bp,
                   const float* __restrict__ bias,
                   const float* __restrict__ agg,
                   const float* __restrict__ deg,
                   float* __restrict__ out) {
    int ntile = blockIdx.x;
    int ctile = threadIdx.x >> 5;
    int lane  = threadIdx.x & 31;
    int rowA  = ntile * 16 + (lane & 15);
    int colD  = ctile * 16 + (lane & 15);
    int rbase = ntile * 16 + (lane >> 4) * 8;

    // issue all A/B loads up front, then the WMMA chain
    v16bf a[4], b[4];
#pragma unroll
    for (int q = 0; q < 4; ++q) {
        a[q] = load_a_frag(hbf, rowA, q, lane);
        b[q] = *(const v16bf*)(Bp + (size_t)((ctile * 4 + q) * 512 + lane * 16));
    }

    v8f acc;
    float bcol = bias[colD];
#pragma unroll
    for (int j = 0; j < 8; ++j) {
        float v = bcol;
        if (agg) {
            float d = fmaxf(deg[rbase + j], 1.0f);
            v += agg[(size_t)(rbase + j) * C + colD] / d;
        }
        acc[j] = v;
    }
#pragma unroll
    for (int q = 0; q < 4; ++q)
        acc = __builtin_amdgcn_wmma_f32_16x16x32_bf16(
                  false, a[q], false, b[q], (short)0, acc, false, false);

#pragma unroll
    for (int j = 0; j < 8; ++j)
        out[(size_t)(rbase + j) * C + colD] = acc[j];
}

// ---------------------------------------------------------------------------
// Edge kernel: one wave per edge; 4 bilinear B-spline taps; each lane carries
// a float4 channel slice (32 lanes * 16 B = full 512 B row per gather), then
// global_atomic_add_f32 scatter into agg + degree count.
// ---------------------------------------------------------------------------
__global__ void edge_kernel(const int* __restrict__ ei,
                            const float* __restrict__ ea,
                            const float* __restrict__ T,
                            float* __restrict__ agg,
                            float* __restrict__ deg) {
    int e = blockIdx.x * 8 + (threadIdx.x >> 5);
    if (e >= N_EDGES) return;
    int lane = threadIdx.x & 31;
    int src = ei[e];
    int dst = ei[N_EDGES + e];

    float v0 = ea[2 * e]     * (float)(KS - 1);
    float v1 = ea[2 * e + 1] * (float)(KS - 1);
    float lo0f = fminf(fmaxf(floorf(v0), 0.0f), (float)(KS - 2));
    float lo1f = fminf(fmaxf(floorf(v1), 0.0f), (float)(KS - 2));
    float f0 = v0 - lo0f, f1 = v1 - lo1f;
    int lo0 = (int)lo0f, lo1 = (int)lo1f;

    const float4* Tb = (const float4*)(T + (size_t)src * (K * C));
    float4 m = make_float4(0.f, 0.f, 0.f, 0.f);
#pragma unroll
    for (int bits = 0; bits < 4; ++bits) {
        int b0 = bits & 1, b1 = (bits >> 1) & 1;
        int idx = (lo0 + b0) + (lo1 + b1) * KS;
        float w = (b0 ? f0 : 1.0f - f0) * (b1 ? f1 : 1.0f - f1);
        float4 t = Tb[idx * (C / 4) + lane];
        m.x += w * t.x; m.y += w * t.y; m.z += w * t.z; m.w += w * t.w;
    }
    float* ag = agg + (size_t)dst * C + lane * 4;
    atomicAdd(ag + 0, m.x);
    atomicAdd(ag + 1, m.y);
    atomicAdd(ag + 2, m.z);
    atomicAdd(ag + 3, m.w);
    if (lane == 0) atomicAdd(deg + dst, 1.0f);
}

// ---------------------------------------------------------------------------
// BatchNorm: per-channel sum/sumsq reduction then normalize(+ReLU), emitting
// the bf16 operand for the next layer's WMMA GEMMs.
// ---------------------------------------------------------------------------
__global__ void bn_stats_kernel(const float* __restrict__ h,
                                float* __restrict__ sums) {
    int c  = threadIdx.x;                 // blockDim.x == 128
    int n0 = blockIdx.x * 128;
    int n1 = n0 + 128; if (n1 > N_NODES) n1 = N_NODES;
    float s = 0.f, s2 = 0.f;
    for (int n = n0; n < n1; ++n) {
        float v = h[(size_t)n * C + c];
        s += v; s2 += v * v;
    }
    atomicAdd(sums + c, s);
    atomicAdd(sums + C + c, s2);
}

__global__ void bn_apply_kernel(const float* __restrict__ h,
                                const float* __restrict__ sums,
                                const float* __restrict__ gamma,
                                const float* __restrict__ beta,
                                int relu,
                                bf16* __restrict__ hbf) {
    int i = blockIdx.x * 256 + threadIdx.x;
    if (i >= N_NODES * C) return;
    int c = i & 127;
    float mu  = sums[c] * (1.0f / N_NODES);
    float var = sums[C + c] * (1.0f / N_NODES) - mu * mu;
    float v = gamma[c] * (h[i] - mu) * rsqrtf(var + BN_EPS) + beta[c];
    if (relu) v = fmaxf(v, 0.0f);
    hbf[i] = (bf16)v;
}

// ---------------------------------------------------------------------------
// Launcher
// ---------------------------------------------------------------------------
extern "C" void kernel_launch(void* const* d_in, const int* in_sizes, int n_in,
                              void* d_out, int out_size, void* d_ws, size_t ws_size,
                              hipStream_t stream) {
    const int*   x    = (const int*)d_in[0];
    const int*   ei   = (const int*)d_in[1];   // [2, E]
    const float* ea   = (const float*)d_in[2]; // [E, 2]
    const float* emb  = (const float*)d_in[3]; // [2, C]
    const float* Wc   = (const float*)d_in[4]; // [3, 25, 128, 128]
    const float* root = (const float*)d_in[5]; // [3, 128, 128]
    const float* bias = (const float*)d_in[6]; // [3, 128]
    const float* gam  = (const float*)d_in[7];
    const float* bet  = (const float*)d_in[8];
    const float* fw   = (const float*)d_in[9]; // [128, 128]
    const float* fb   = (const float*)d_in[10];
    float* out = (float*)d_out;

    char* ws = (char*)d_ws;
    size_t cur = 0;
    auto carve = [&](size_t bytes) -> char* {
        char* p = ws + cur;
        cur = (cur + bytes + 255) & ~(size_t)255;
        return p;
    };
    float* T    = (float*)carve((size_t)N_NODES * K * C * sizeof(float)); // 640 MB
    float* hpre = (float*)carve((size_t)N_NODES * C * sizeof(float));
    float* agg  = (float*)carve((size_t)N_NODES * C * sizeof(float));
    float* deg  = (float*)carve((size_t)N_NODES * sizeof(float));
    float* sums = (float*)carve(2 * C * sizeof(float));
    bf16*  hbf  = (bf16*) carve((size_t)N_NODES * C * sizeof(bf16));
    bf16*  Wp   = (bf16*) carve((size_t)79 * 16384 * sizeof(bf16));
    bf16* Wp_conv = Wp;                        // 75 matrices (3 layers x 25)
    bf16* Wp_root = Wp + (size_t)75 * 16384;   // 3 matrices
    bf16* Wp_fin  = Wp + (size_t)78 * 16384;   // 1 matrix (transposed)

    pack_b_kernel<<<(75 * 16384 + 255) / 256, 256, 0, stream>>>(Wc,   Wp_conv, 75, 0);
    pack_b_kernel<<<( 3 * 16384 + 255) / 256, 256, 0, stream>>>(root, Wp_root,  3, 0);
    pack_b_kernel<<<(     16384 + 255) / 256, 256, 0, stream>>>(fw,   Wp_fin,   1, 1);

    embed_kernel<<<(N_NODES * C + 255) / 256, 256, 0, stream>>>(x, emb, hbf);

    for (int L = 0; L < 3; ++L) {
        gemm_transform_kernel<<<NTILES, 256, 0, stream>>>(
            hbf, Wp_conv + (size_t)L * 25 * 16384, T);

        hipMemsetAsync(agg, 0, (size_t)N_NODES * C * sizeof(float), stream);
        hipMemsetAsync(deg, 0, (size_t)N_NODES * sizeof(float), stream);
        edge_kernel<<<(N_EDGES + 7) / 8, 256, 0, stream>>>(ei, ea, T, agg, deg);

        gemm_rowlin_kernel<<<NTILES, 256, 0, stream>>>(
            hbf, Wp_root + (size_t)L * 16384, bias + L * C, agg, deg, hpre);

        hipMemsetAsync(sums, 0, 2 * C * sizeof(float), stream);
        bn_stats_kernel<<<(N_NODES + 127) / 128, 128, 0, stream>>>(hpre, sums);
        bn_apply_kernel<<<(N_NODES * C + 255) / 256, 256, 0, stream>>>(
            hpre, sums, gam + L * C, bet + L * C, (L != 2) ? 1 : 0, hbf);
    }

    gemm_rowlin_kernel<<<NTILES, 256, 0, stream>>>(
        hbf, Wp_fin, fb, nullptr, nullptr, out);

    (void)in_sizes; (void)n_in; (void)out_size; (void)ws_size;
}
